// LlamaAttention_52012053954983
// MI455X (gfx1250) — matrix operand order repaired and checked by reading, not compiled, not run on hardware
//
#include <hip/hip_runtime.h>
#include <hip/hip_bf16.h>

typedef __attribute__((ext_vector_type(16))) _Float16 v16h;
typedef __attribute__((ext_vector_type(8)))  _Float16 v8h;
typedef __attribute__((ext_vector_type(4)))  _Float16 v4h;
typedef __attribute__((ext_vector_type(8)))  float    v8f;
typedef __attribute__((ext_vector_type(4)))  int      v4i;

static constexpr int Bc  = 2;
static constexpr int Sq  = 2048;
static constexpr int Dm  = 2048;
static constexpr int QH  = 16;
static constexpr int KVH = 4;
static constexpr int HD  = 128;

#define AS_GLOBAL __attribute__((address_space(1)))
#define AS_LDS    __attribute__((address_space(3)))

#if defined(__gfx1250__) && __has_builtin(__builtin_amdgcn_global_load_async_to_lds_b128)
#define HAVE_ASYNC_LDS 1
#else
#define HAVE_ASYNC_LDS 0
#endif

// 16B global -> LDS copy. Async (ASYNCcnt-tracked, no VGPR staging) when available.
__device__ inline void cp_async16(_Float16* lds, const _Float16* g) {
#if HAVE_ASYNC_LDS
  __builtin_amdgcn_global_load_async_to_lds_b128(
      (AS_GLOBAL v4i*)g, (AS_LDS v4i*)lds, 0, 0);
#else
  *(v8h*)lds = *(const v8h*)g;
#endif
}
__device__ inline void wait_async_all() {
#if HAVE_ASYNC_LDS
  asm volatile("s_wait_asynccnt 0x0" ::: "memory");
#endif
}

// ---------------------------------------------------------------------------
// WMMA operand loaders following CDNA5 ISA §7.12.2 VGPR layouts (wave32).
// A (16x32 f16): lane r=lane&15 holds row r; k-chunks [8*(lane>>4), +8) and
//                [16+8*(lane>>4), +8).
// B (32x16 f16): lane n=lane&15 holds column n; k-chunks [8*(lane>>4), +8) and
//                [16+8*(lane>>4), +8). A row-major [n][k] source needs no transpose.
// ---------------------------------------------------------------------------
__device__ inline v16h ld_a16x32(const _Float16* base, int stride) {
  const int lane = threadIdx.x & 31;
  const int r  = lane & 15;
  const int kb = (lane >> 4) << 3;
  const _Float16* p = base + r * stride + kb;
  v8h lo = *(const v8h*)(p);
  v8h hi = *(const v8h*)(p + 16);
  return __builtin_shufflevector(lo, hi, 0,1,2,3,4,5,6,7,8,9,10,11,12,13,14,15);
}

__device__ inline v16h ld_b32x16(const _Float16* base, int stride) {
  const int lane = threadIdx.x & 31;
  const int n  = lane & 15;
  const int kb = (lane >> 4) << 3;
  const _Float16* p = base + n * stride + kb;
  v8h lo = *(const v8h*)(p);
  v8h hi = *(const v8h*)(p + 16);
  return __builtin_shufflevector(lo, hi, 0,1,2,3,4,5,6,7,8,9,10,11,12,13,14,15);
}

// ---------------------------------------------------------------------------
// fp32 -> f16 convert (vec4)
// ---------------------------------------------------------------------------
__global__ void cvt_f32_f16(const float* __restrict__ in, _Float16* __restrict__ out, int n4) {
  int i = blockIdx.x * blockDim.x + threadIdx.x;
  if (i >= n4) return;
  float4 v = ((const float4*)in)[i];
  v4h o;
  o[0] = (_Float16)v.x; o[1] = (_Float16)v.y; o[2] = (_Float16)v.z; o[3] = (_Float16)v.w;
  *(v4h*)&out[(size_t)i * 4] = o;
}

// ---------------------------------------------------------------------------
// GEMM: out[M,N] = A[M,K] @ W[N,K]^T    (f16 in, f32 accum, OutT out)
// Block 128x128, BK=32, 8 waves (2x4), each wave 64x32 = 4x2 WMMA C-tiles.
// Double-buffered LDS, async global->LDS staging, one barrier per K-slab.
// ---------------------------------------------------------------------------
template <typename OutT>
__global__ __launch_bounds__(256) void gemm_xwT(const _Float16* __restrict__ A,
                                                const _Float16* __restrict__ W,
                                                OutT* __restrict__ out,
                                                int N, int K) {
  constexpr int LDT = 40;  // 32 + 8 pad
  __shared__ __align__(16) _Float16 As[2][128 * LDT];
  __shared__ __align__(16) _Float16 Bs[2][128 * LDT];

  const int t = threadIdx.x;
  const int lane = t & 31;
  const int w  = t >> 5;
  const int wm = w >> 2;   // 0..1
  const int wn = w & 3;    // 0..3
  const int m0 = blockIdx.y * 128;
  const int n0 = blockIdx.x * 128;

  auto stage = [&](int buf, int k0) {
#pragma unroll
    for (int q = 0; q < 2; ++q) {
      int ch  = t * 2 + q;            // 0..511
      int row = ch >> 2;
      int cc  = (ch & 3) * 8;
      cp_async16(&As[buf][row * LDT + cc], &A[(size_t)(m0 + row) * K + k0 + cc]);
      cp_async16(&Bs[buf][row * LDT + cc], &W[(size_t)(n0 + row) * K + k0 + cc]);
    }
  };

  v8f c[4][2];
#pragma unroll
  for (int i = 0; i < 4; ++i)
#pragma unroll
    for (int j = 0; j < 2; ++j)
#pragma unroll
      for (int r = 0; r < 8; ++r) c[i][j][r] = 0.0f;

  stage(0, 0);
  const int nk = K / 32;
  for (int ki = 0; ki < nk; ++ki) {
    wait_async_all();
    __syncthreads();                       // buf[ki&1] ready on all waves
    if (ki + 1 < nk) stage((ki + 1) & 1, (ki + 1) * 32);
    if (ki + 2 < nk)                       // L2 prefetch two slabs ahead
      __builtin_prefetch(&A[(size_t)(m0 + (t & 127)) * K + (ki + 2) * 32 + (t >> 7) * 16], 0, 1);

    const _Float16* as = As[ki & 1];
    const _Float16* bs = Bs[ki & 1];
    v16h bt[2], at[4];
#pragma unroll
    for (int j = 0; j < 2; ++j)
      bt[j] = ld_b32x16(&bs[(wn * 32 + j * 16) * LDT], LDT);
#pragma unroll
    for (int i = 0; i < 4; ++i)
      at[i] = ld_a16x32(&as[(wm * 64 + i * 16) * LDT], LDT);
#pragma unroll
    for (int i = 0; i < 4; ++i)
#pragma unroll
      for (int j = 0; j < 2; ++j)
        c[i][j] = __builtin_amdgcn_wmma_f32_16x16x32_f16(
            false, at[i], false, bt[j], (short)0, c[i][j], false, false);
  }

  const int half = lane >> 4;
  const int col  = lane & 15;
#pragma unroll
  for (int i = 0; i < 4; ++i)
#pragma unroll
    for (int j = 0; j < 2; ++j)
#pragma unroll
      for (int r = 0; r < 8; ++r) {
        int row  = m0 + wm * 64 + i * 16 + r + 8 * half;
        int ccol = n0 + wn * 32 + j * 16 + col;
        out[(size_t)row * N + ccol] = (OutT)c[i][j][r];
      }
}

// ---------------------------------------------------------------------------
// RoPE (rotate-half, base 500000) in-place on f16 [B][S][H][HD]
// ---------------------------------------------------------------------------
__global__ void rope_kernel(_Float16* __restrict__ tbuf, int H, int n) {
  int tid = blockIdx.x * blockDim.x + threadIdx.x;
  if (tid >= n) return;
  int pair = tid & 63;
  int rest = tid >> 6;
  int h = rest % H;  rest /= H;
  int s = rest % Sq;
  int b = rest / Sq;
  size_t base = (((size_t)b * Sq + s) * H + h) * HD;
  float v1 = (float)tbuf[base + pair];
  float v2 = (float)tbuf[base + pair + 64];
  float freq = __expf(-(float)pair * 0.20503692776459218f);  // BASE^(-pair/64)
  float ang = (float)s * freq;
  float cs = __cosf(ang), sn = __sinf(ang);
  tbuf[base + pair]      = (_Float16)(v1 * cs - v2 * sn);
  tbuf[base + pair + 64] = (_Float16)(v2 * cs + v1 * sn);
}

// ---------------------------------------------------------------------------
// V transpose: v[b][s][h][d] -> vt[b][h][d][s]
// ---------------------------------------------------------------------------
__global__ void vtrans_kernel(const _Float16* __restrict__ v, _Float16* __restrict__ vt, int n) {
  int tid = blockIdx.x * blockDim.x + threadIdx.x;
  if (tid >= n) return;
  int s = tid % Sq;  int rest = tid / Sq;
  int d = rest % HD; rest /= HD;
  int h = rest % KVH;
  int b = rest / KVH;
  vt[tid] = v[(((size_t)b * Sq + s) * KVH + h) * HD + d];
}

// ---------------------------------------------------------------------------
// Flash attention (causal, GQA). Grid: (S/128, QH, B); 256 threads = 8 waves,
// each wave owns 16 query rows. K-block = 32 keys. Double-buffered K/V tiles.
// ---------------------------------------------------------------------------
__global__ __launch_bounds__(256) void attn_fwd(const _Float16* __restrict__ Q,   // [B][S][QH][HD]
                                                const _Float16* __restrict__ Kt,  // [B][S][KVH][HD]
                                                const _Float16* __restrict__ Vt,  // [B][KVH][HD][S]
                                                _Float16* __restrict__ O) {       // [B][S][QH][HD]
  constexpr int LDK = 136;  // 128 + 8 pad
  constexpr int LDV = 40;   // 32 + 8 pad
  constexpr int LDP = 40;
  __shared__ __align__(16) _Float16 Ks[2][32 * LDK];
  __shared__ __align__(16) _Float16 Vs[2][128 * LDV];
  __shared__ __align__(16) _Float16 Ps[8 * 16 * LDP];

  const int t = threadIdx.x, lane = t & 31, w = t >> 5;
  const int qb = blockIdx.x, h = blockIdx.y, b = blockIdx.z;
  const int kvh = h >> 2;                 // QH/KVH = 4
  const int q0 = qb * 128 + w * 16;
  const int half = lane >> 4, col = lane & 15;
  const float scale = 0.08838834764831845f;  // 1/sqrt(128)

  const _Float16* qbase  = Q  + (((size_t)b * Sq + q0) * QH + h) * HD;
  const _Float16* kbaseg = Kt + (((size_t)b * Sq) * KVH + kvh) * HD;         // + s*KVH*HD
  const _Float16* vbaseg = Vt + ((size_t)b * KVH + kvh) * (size_t)HD * Sq;   // + d*S + s

  auto stage = [&](int buf, int kb) {
#pragma unroll
    for (int q = 0; q < 2; ++q) {
      int ch = t * 2 + q;                 // 0..511
      int krow = ch >> 4, kc = (ch & 15) * 8;
      cp_async16(&Ks[buf][krow * LDK + kc],
                 &kbaseg[(size_t)(kb * 32 + krow) * (KVH * HD) + kc]);
      int vrow = ch >> 2, vc = (ch & 3) * 8;
      cp_async16(&Vs[buf][vrow * LDV + vc],
                 &vbaseg[(size_t)vrow * Sq + kb * 32 + vc]);
    }
  };

  v16h aq[4];
#pragma unroll
  for (int j = 0; j < 4; ++j) aq[j] = ld_a16x32(qbase + j * 32, QH * HD);

  v8f o[8];
  float m_i[8], l_i[8];
#pragma unroll
  for (int r = 0; r < 8; ++r) { m_i[r] = -3.0e38f; l_i[r] = 0.0f; }
#pragma unroll
  for (int dt = 0; dt < 8; ++dt)
#pragma unroll
    for (int r = 0; r < 8; ++r) o[dt][r] = 0.0f;

  const int nkb    = qb * 4 + 4;           // WG-level causal bound
  const int my_nkb = ((q0 + 15) >> 5) + 1; // this wave's causal bound (uniform per wave)

  stage(0, 0);
  for (int kb = 0; kb < nkb; ++kb) {
    wait_async_all();
    __syncthreads();                      // buf[kb&1] ready on all waves
    if (kb + 1 < nkb) stage((kb + 1) & 1, kb + 1);

    if (kb < my_nkb) {                    // skip fully-masked blocks (wave-uniform)
      const _Float16* ks = Ks[kb & 1];
      const _Float16* vs = Vs[kb & 1];

      // scores: S = Q @ K^T  (two 16x16 n-tiles over 32 keys, K-dim = 128)
      v8f sc[2];
#pragma unroll
      for (int nt = 0; nt < 2; ++nt) {
        v16h bk[4];
#pragma unroll
        for (int j = 0; j < 4; ++j)
          bk[j] = ld_b32x16(&ks[(nt * 16) * LDK + j * 32], LDK);
        v8f acc;
#pragma unroll
        for (int r = 0; r < 8; ++r) acc[r] = 0.0f;
#pragma unroll
        for (int j = 0; j < 4; ++j)
          acc = __builtin_amdgcn_wmma_f32_16x16x32_f16(
              false, aq[j], false, bk[j], (short)0, acc, false, false);
        sc[nt] = acc;
      }

      // scale + causal mask (C layout: row = r + 8*half, col = nt*16 + col)
#pragma unroll
      for (int nt = 0; nt < 2; ++nt)
#pragma unroll
        for (int r = 0; r < 8; ++r) {
          int key  = kb * 32 + nt * 16 + col;
          int qrow = q0 + r + 8 * half;
          sc[nt][r] = (key <= qrow) ? sc[nt][r] * scale : -3.0e38f;
        }

      // online softmax (reductions within 16-lane halves match C layout)
      float alpha[8];
#pragma unroll
      for (int r = 0; r < 8; ++r) {
        float v = fmaxf(sc[0][r], sc[1][r]);
        v = fmaxf(v, __shfl_xor(v, 1, 32));
        v = fmaxf(v, __shfl_xor(v, 2, 32));
        v = fmaxf(v, __shfl_xor(v, 4, 32));
        v = fmaxf(v, __shfl_xor(v, 8, 32));
        float mn = fmaxf(m_i[r], v);
        alpha[r] = __expf(m_i[r] - mn);
        m_i[r] = mn;
      }
#pragma unroll
      for (int nt = 0; nt < 2; ++nt)
#pragma unroll
        for (int r = 0; r < 8; ++r) sc[nt][r] = __expf(sc[nt][r] - m_i[r]);
#pragma unroll
      for (int r = 0; r < 8; ++r) {
        float rs = sc[0][r] + sc[1][r];
        rs += __shfl_xor(rs, 1, 32);
        rs += __shfl_xor(rs, 2, 32);
        rs += __shfl_xor(rs, 4, 32);
        rs += __shfl_xor(rs, 8, 32);
        l_i[r] = l_i[r] * alpha[r] + rs;
      }
#pragma unroll
      for (int dt = 0; dt < 8; ++dt)
#pragma unroll
        for (int r = 0; r < 8; ++r) o[dt][r] *= alpha[r];

      // P (C-layout f32) -> per-wave LDS -> A-layout f16
      _Float16* pb = &Ps[w * 16 * LDP];
#pragma unroll
      for (int nt = 0; nt < 2; ++nt)
#pragma unroll
        for (int r = 0; r < 8; ++r)
          pb[(r + 8 * half) * LDP + nt * 16 + col] = (_Float16)sc[nt][r];
      asm volatile("s_wait_dscnt 0x0" ::: "memory");  // intra-wave store->load ordering
      v16h ap = ld_a16x32(pb, LDP);

      // O += P(16x32) @ V(32x128): 8 d-tiles, grouped loads then WMMA chain
#pragma unroll
      for (int g = 0; g < 2; ++g) {
        v16h bv[4];
#pragma unroll
        for (int d = 0; d < 4; ++d)
          bv[d] = ld_b32x16(&vs[((g * 4 + d) * 16) * LDV], LDV);
#pragma unroll
        for (int d = 0; d < 4; ++d)
          o[g * 4 + d] = __builtin_amdgcn_wmma_f32_16x16x32_f16(
              false, ap, false, bv[d], (short)0, o[g * 4 + d], false, false);
      }
    }
  }

  // normalize + store f16
  _Float16* obase = O + (((size_t)b * Sq + q0) * QH + h) * HD;
#pragma unroll
  for (int r = 0; r < 8; ++r) {
    float inv = 1.0f / l_i[r];
#pragma unroll
    for (int dt = 0; dt < 8; ++dt)
      obase[(size_t)(r + 8 * half) * (QH * HD) + dt * 16 + col] =
          (_Float16)(o[dt][r] * inv);
  }
}

// ---------------------------------------------------------------------------
// Launch
// ---------------------------------------------------------------------------
extern "C" void kernel_launch(void* const* d_in, const int* in_sizes, int n_in,
                              void* d_out, int out_size, void* d_ws, size_t ws_size,
                              hipStream_t stream) {
  const float* x  = (const float*)d_in[0];
  const float* wq = (const float*)d_in[1];
  const float* wk = (const float*)d_in[2];
  const float* wv = (const float*)d_in[3];
  const float* wo = (const float*)d_in[4];
  float* out = (float*)d_out;

  const size_t BS = (size_t)Bc * Sq;          // 4096 tokens
  const size_t nX  = BS * Dm;                 // 8388608
  const size_t nWq = (size_t)Dm * Dm;         // 4194304
  const size_t nWk = (size_t)(KVH * HD) * Dm; // 1048576
  const size_t nKV = BS * KVH * HD;           // 2097152

  char* p = (char*)d_ws;
  auto take = [&](size_t elems) { _Float16* r = (_Float16*)p; p += elems * sizeof(_Float16); return r; };
  _Float16* xh  = take(nX);
  _Float16* wqh = take(nWq);
  _Float16* wkh = take(nWk);
  _Float16* wvh = take(nWk);
  _Float16* woh = take(nWq);
  _Float16* qh  = take(nX);    // q proj (then RoPE'd)
  _Float16* kh  = take(nKV);   // k proj (then RoPE'd)
  _Float16* vh  = take(nKV);
  _Float16* vth = take(nKV);   // V transposed [b][h][d][s]
  _Float16* ah  = take(nX);    // attention output

  // 1) fp32 -> f16
  cvt_f32_f16<<<(nX  / 4 + 255) / 256, 256, 0, stream>>>(x,  xh,  (int)(nX  / 4));
  cvt_f32_f16<<<(nWq / 4 + 255) / 256, 256, 0, stream>>>(wq, wqh, (int)(nWq / 4));
  cvt_f32_f16<<<(nWk / 4 + 255) / 256, 256, 0, stream>>>(wk, wkh, (int)(nWk / 4));
  cvt_f32_f16<<<(nWk / 4 + 255) / 256, 256, 0, stream>>>(wv, wvh, (int)(nWk / 4));
  cvt_f32_f16<<<(nWq / 4 + 255) / 256, 256, 0, stream>>>(wo, woh, (int)(nWq / 4));

  // 2) projections: q = x@wq^T, k = x@wk^T, v = x@wv^T
  gemm_xwT<_Float16><<<dim3(Dm / 128, (int)(BS / 128)), 256, 0, stream>>>(xh, wqh, qh, Dm, Dm);
  gemm_xwT<_Float16><<<dim3(KVH * HD / 128, (int)(BS / 128)), 256, 0, stream>>>(xh, wkh, kh, KVH * HD, Dm);
  gemm_xwT<_Float16><<<dim3(KVH * HD / 128, (int)(BS / 128)), 256, 0, stream>>>(xh, wvh, vh, KVH * HD, Dm);

  // 3) RoPE on q and k
  {
    int nq = (int)(BS * QH * 64);
    int nk = (int)(BS * KVH * 64);
    rope_kernel<<<(nq + 255) / 256, 256, 0, stream>>>(qh, QH, nq);
    rope_kernel<<<(nk + 255) / 256, 256, 0, stream>>>(kh, KVH, nk);
  }

  // 4) V transpose for PV B-operand
  vtrans_kernel<<<((int)nKV + 255) / 256, 256, 0, stream>>>(vh, vth, (int)nKV);

  // 5) flash attention
  attn_fwd<<<dim3(Sq / 128, QH, Bc), 256, 0, stream>>>(qh, kh, vth, ah);

  // 6) output projection (f32 out)
  gemm_xwT<float><<<dim3(Dm / 128, (int)(BS / 128)), 256, 0, stream>>>(ah, woh, out, Dm, Dm);
}
